// TrajectoryDecoder_86053964743009
// MI455X (gfx1250) — compile-verified, matrix operand order checked by reading, not run on hardware
//
#include <hip/hip_runtime.h>
#include <math.h>

// ---------------- problem constants ----------------
static constexpr int kNL = 6, kNQ = 128, kFD = 256, kHD = 512, kNH = 8;
static constexpr int kHd = kHD / kNH;              // 64
static constexpr int kNC = 6, kH = 48, kW = 120, kHW = kH * kW;
static constexpr int kP  = 96;                     // points per query
static constexpr int kNOUT = 24;                   // 11 + 10 + 3

// ---------------- WMMA types ----------------
typedef __attribute__((ext_vector_type(16))) __bf16 v16bf;
typedef __attribute__((ext_vector_type(8)))  float  v8f;

union FragU { unsigned u[8]; v16bf bf; };

// fp32 -> bf16 bits, round-to-nearest-even
__device__ __forceinline__ unsigned short bf_bits(float f) {
  union { float f; unsigned u; } v; v.f = f;
  return (unsigned short)((v.u + 0x7FFFu + ((v.u >> 16) & 1u)) >> 16);
}
// pack two fp32 -> dword of 2x bf16 (lo in [15:0])
__device__ __forceinline__ unsigned pk_bf16(float lo, float hi) {
  union { float f; unsigned u; } a, b;
  a.f = lo; b.f = hi;
  unsigned ra = (a.u + 0x7FFFu + ((a.u >> 16) & 1u)) >> 16;
  unsigned rb = (b.u + 0x7FFFu + ((b.u >> 16) & 1u)) & 0xFFFF0000u;
  return ra | rb;
}

// ---------------- bulk fp32 -> bf16 conversion (8 elems / thread) ----------------
__global__ void cvt_bf16_kernel(const float* __restrict__ in,
                                unsigned short* __restrict__ out, long n8) {
  long t = (long)blockIdx.x * blockDim.x + threadIdx.x;
  if (t >= n8) return;
  const float4* p = (const float4*)(in + t * 8);
  float4 a = p[0], b = p[1];
  uint4 o;
  o.x = pk_bf16(a.x, a.y); o.y = pk_bf16(a.z, a.w);
  o.z = pk_bf16(b.x, b.y); o.w = pk_bf16(b.z, b.w);
  *(uint4*)(out + t * 8) = o;
}

// Per-operand-tile fragment descriptor: lane-resolved row pointer + zero mask.
// CDNA5 16-bit A layout: lanes 0-15 hold i=lane (kb=0), lanes 16-31 hold
// i=lane-16 (kb=8); per lane two contiguous 8-bf16 runs per 32-K step.
struct Frag { const unsigned short* row; unsigned msk; };

__device__ __forceinline__ Frag mk_frag(const unsigned short* base, int i0,
                                        long si, int imax, bool bound) {
  int lane = threadIdx.x & 31;
  int ii = i0 + (lane & 15);
  int kb = (lane >> 4) << 3;           // 0 or 8
  Frag f;
  if (bound) {
    bool iok = (ii < imax);
    int i = iok ? ii : (imax - 1);
    f.row = base + (long)i * si + kb;
    f.msk = iok ? 0xFFFFFFFFu : 0u;
  } else {
    f.row = base + (long)ii * si + kb;
    f.msk = 0xFFFFFFFFu;
  }
  return f;
}

template <bool BOUND>
__device__ __forceinline__ v16bf ld_frag(const Frag& f, int k0) {
  uint4 c0 = *(const uint4*)(f.row + k0);        // k = kb+0..7
  uint4 c1 = *(const uint4*)(f.row + k0 + 16);   // k = 16+kb+0..7
  FragU r;
  if (BOUND) {
    r.u[0] = c0.x & f.msk; r.u[1] = c0.y & f.msk;
    r.u[2] = c0.z & f.msk; r.u[3] = c0.w & f.msk;
    r.u[4] = c1.x & f.msk; r.u[5] = c1.y & f.msk;
    r.u[6] = c1.z & f.msk; r.u[7] = c1.w & f.msk;
  } else {
    r.u[0] = c0.x; r.u[1] = c0.y; r.u[2] = c0.z; r.u[3] = c0.w;
    r.u[4] = c1.x; r.u[5] = c1.y; r.u[6] = c1.z; r.u[7] = c1.w;
  }
  return r.bf;
}

// Generic WMMA GEMM on pre-converted bf16 operands.
// C[m,n] = act( sum_k A[m,k]*B[n,k] + bias[n] ) (+range)(+resid)
// Writes optional fp32 output Cf and/or bf16 output C16.
// blockIdx.z adds z-strides (attention heads). One wave per block computes a
// 16(M) x 64(N) tile. K % 32 == 0 and M % 16 == 0 required.
// BOUND=true only for N not a multiple of 16 (prediction-head outputs).
template <bool BOUND>
__global__ __launch_bounds__(32)
void gemm_wmma_bf16(const unsigned short* __restrict__ A, long a_si, long a_zs,
                    const unsigned short* __restrict__ Bm, long b_si, long b_zs,
                    const float* __restrict__ bias,
                    const float* __restrict__ resid, long r_ld, long r_zs,
                    float* __restrict__ Cf, long c_ld, long c_zs,
                    unsigned short* __restrict__ C16, long c16_ld, long c16_zs,
                    int M, int N, int K, int act,
                    const float* __restrict__ pmin, const float* __restrict__ pmax) {
  int z = blockIdx.z;
  A  += (long)z * a_zs;
  Bm += (long)z * b_zs;
  if (Cf)    Cf    += (long)z * c_zs;
  if (C16)   C16   += (long)z * c16_zs;
  if (resid) resid += (long)z * r_zs;

  int m0 = blockIdx.y * 16;
  int n0 = blockIdx.x * 64;

  Frag fa  = mk_frag(A,  m0,      a_si, M, false);
  Frag fb0 = mk_frag(Bm, n0,      b_si, N, BOUND);
  Frag fb1 = mk_frag(Bm, n0 + 16, b_si, N, BOUND);
  Frag fb2 = mk_frag(Bm, n0 + 32, b_si, N, BOUND);
  Frag fb3 = mk_frag(Bm, n0 + 48, b_si, N, BOUND);

  v8f acc0 = {}, acc1 = {}, acc2 = {}, acc3 = {};
  for (int k0 = 0; k0 < K; k0 += 32) {
    v16bf a  = ld_frag<false>(fa,  k0);
    v16bf b0 = ld_frag<BOUND>(fb0, k0);
    v16bf b1 = ld_frag<BOUND>(fb1, k0);
    v16bf b2 = ld_frag<BOUND>(fb2, k0);
    v16bf b3 = ld_frag<BOUND>(fb3, k0);
    acc0 = __builtin_amdgcn_wmma_f32_16x16x32_bf16(false, a, false, b0, (short)0, acc0, false, false);
    acc1 = __builtin_amdgcn_wmma_f32_16x16x32_bf16(false, a, false, b1, (short)0, acc1, false, false);
    acc2 = __builtin_amdgcn_wmma_f32_16x16x32_bf16(false, a, false, b2, (short)0, acc2, false, false);
    acc3 = __builtin_amdgcn_wmma_f32_16x16x32_bf16(false, a, false, b3, (short)0, acc3, false, false);
  }

  // C/D layout: lane&15 = column, (lane>>4)*8 + vgpr = row
  int lane = threadIdx.x & 31;
  int nn = lane & 15;
  int mb = (lane >> 4) << 3;
  v8f accs[4] = {acc0, acc1, acc2, acc3};
#pragma unroll
  for (int t = 0; t < 4; ++t) {
    int n = n0 + t * 16 + nn;
    if (n >= N) continue;
    float bv = bias ? bias[n] : 0.0f;
    float rmin = 0.0f, rscale = 1.0f;
    if (pmin) { rmin = pmin[n]; rscale = pmax[n] - pmin[n]; }
#pragma unroll
    for (int v = 0; v < 8; ++v) {
      int m = m0 + mb + v;
      float val = accs[t][v] + bv;
      if (act == 1)      val = fmaxf(val, 0.0f);
      else if (act == 2) val = 1.0f / (1.0f + expf(-val));
      if (pmin)  val = val * rscale + rmin;
      if (resid) val += resid[(long)m * r_ld + n];
      if (Cf)  Cf[(long)m * c_ld + n] = val;
      if (C16) C16[(long)m * c16_ld + n] = bf_bits(val);
    }
  }
}

// ---------------- V-transpose (bf16 -> bf16): vT[h*64+n][k] = qkv16[k][1024+h*64+n]
__global__ void vT_kernel(const unsigned short* __restrict__ qkv16,
                          unsigned short* __restrict__ vT) {
  int t = blockIdx.x * blockDim.x + threadIdx.x;  // 512*128
  if (t >= kHD * kNQ) return;
  int r = t >> 7;        // 0..511
  int c = t & 127;       // 0..127
  vT[(long)r * kNQ + c] = qkv16[(long)c * (3 * kHD) + 2 * kHD + r];
}

// ---------------- LayerNorm: one block (256 thr) per row of 512 ----------------
__global__ void layernorm512(const float* __restrict__ in, float* __restrict__ out,
                             unsigned short* __restrict__ out16,
                             const float* __restrict__ g, const float* __restrict__ b) {
  int row = blockIdx.x;
  const float* x = in + (long)row * kHD;
  __shared__ float red[256];
  int t = threadIdx.x;
  float v0 = x[t], v1 = x[t + 256];
  red[t] = v0 + v1; __syncthreads();
  for (int o = 128; o > 0; o >>= 1) { if (t < o) red[t] += red[t + o]; __syncthreads(); }
  float mean = red[0] * (1.0f / kHD); __syncthreads();
  float d0 = v0 - mean, d1 = v1 - mean;
  red[t] = d0 * d0 + d1 * d1; __syncthreads();
  for (int o = 128; o > 0; o >>= 1) { if (t < o) red[t] += red[t + o]; __syncthreads(); }
  float inv = rsqrtf(red[0] * (1.0f / kHD) + 1e-5f);
  float y0 = d0 * inv * g[t] + b[t];
  float y1 = d1 * inv * g[t + 256] + b[t + 256];
  out[(long)row * kHD + t]         = y0;
  out[(long)row * kHD + t + 256]   = y1;
  out16[(long)row * kHD + t]       = bf_bits(y0);
  out16[(long)row * kHD + t + 256] = bf_bits(y1);
}

// ---------------- softmax over rows of 128; bf16 result ----------------
__global__ void softmax128(const float* __restrict__ s, unsigned short* __restrict__ o16,
                           float scale) {
  int row = blockIdx.x;
  const float* p = s + (long)row * kNQ;
  __shared__ float red[128];
  int t = threadIdx.x;
  float v = p[t] * scale;
  red[t] = v; __syncthreads();
  for (int o = 64; o > 0; o >>= 1) { if (t < o) red[t] = fmaxf(red[t], red[t + o]); __syncthreads(); }
  float mx = red[0]; __syncthreads();
  float e = expf(v - mx);
  red[t] = e; __syncthreads();
  for (int o = 64; o > 0; o >>= 1) { if (t < o) red[t] += red[t + o]; __syncthreads(); }
  o16[(long)row * kNQ + t] = bf_bits(e / red[0]);
}

// ---------------- box point sampler (from motion slice of preds) ----------------
__global__ void sample_pts_kernel(const float* __restrict__ predsL, float* __restrict__ pts) {
  int t = blockIdx.x * blockDim.x + threadIdx.x;
  if (t >= kNQ * kP) return;
  int n = t / kP, p = t % kP;
  const float* m = predsL + (long)n * kNOUT;
  float cx = m[0], cyq = m[1], cz = m[2];
  float yaw = m[7], dx = m[8], dy = m[9], dz = m[10];
  int face = p >> 4, axis = face >> 1;
  float s = (face & 1) ? 0.5f : -0.5f;
  int i = (p >> 2) & 3, j = p & 3;
  float offi = -0.4f + (0.8f / 3.0f) * i;
  float offj = -0.4f + (0.8f / 3.0f) * j;
  int rem0 = (axis == 0) ? 1 : 0;
  int rem1 = (axis == 2) ? 1 : 2;
  float q[3];
  q[axis] = s; q[rem0] = offi; q[rem1] = offj;
  float px = q[0] * dx, py = q[1] * dy, pz = q[2] * dz;
  float cy = cosf(yaw), sy = sinf(yaw);
  pts[t * 3 + 0] = px * cy - py * sy + cx;
  pts[t * 3 + 1] = px * sy + py * cy + cyq;
  pts[t * 3 + 2] = pz + cz;
}

// ---------------- bilinear gather-aggregate: block=256 (fd), grid=NQ ----------------
__global__ void gather_agg_kernel(const float* __restrict__ feat,   // (NC,FD,H,W)
                                  const float* __restrict__ proj,   // (NC,3,4)
                                  const float* __restrict__ pts,    // (NQ,P,3)
                                  unsigned short* __restrict__ agg16) { // (NQ,FD)
  int n = blockIdx.x;
  int fd = threadIdx.x;
  __shared__ float Pm[kNC * 12];
  if (threadIdx.x < kNC * 12) Pm[threadIdx.x] = proj[threadIdx.x];
  __syncthreads();
  float acc = 0.0f;
  for (int p = 0; p < kP; ++p) {
    const float* pt = pts + ((long)n * kP + p) * 3;
    float x = pt[0], y = pt[1], z3 = pt[2];
#pragma unroll
    for (int c = 0; c < kNC; ++c) {
      const float* Pc = Pm + c * 12;
      float cx = Pc[0] * x + Pc[1] * y + Pc[2]  * z3 + Pc[3];
      float cy = Pc[4] * x + Pc[5] * y + Pc[6]  * z3 + Pc[7];
      float cz = Pc[8] * x + Pc[9] * y + Pc[10] * z3 + Pc[11];
      bool front = cz > 0.0f;
      float zs = (fabsf(cz) > 1e-6f) ? cz : 1e-6f;
      float u = cx / zs, v = cy / zs;
      float u0 = floorf(u), v0 = floorf(v);
      float du = u - u0, dv = v - v0;
      const float* fc = feat + ((long)c * kFD + fd) * kHW;
      float uu[4] = {u0, u0 + 1.0f, u0, u0 + 1.0f};
      float vv[4] = {v0, v0, v0 + 1.0f, v0 + 1.0f};
      float ww[4] = {(1 - du) * (1 - dv), du * (1 - dv), (1 - du) * dv, du * dv};
#pragma unroll
      for (int tp = 0; tp < 4; ++tp) {
        float ui = uu[tp], vi = vv[tp];
        if (front && ui >= 0.0f && ui <= (float)(kW - 1) && vi >= 0.0f && vi <= (float)(kH - 1)) {
          int idx = (int)vi * kW + (int)ui;
          acc += fc[idx] * ww[tp];
        }
      }
    }
  }
  agg16[(long)n * kFD + fd] = bf_bits(acc * (1.0f / (float)(kNC * kP)));
}

// ---------------- host-side helpers ----------------
static inline void cvt(hipStream_t s, const float* in, unsigned short* out, long count) {
  long n8 = count / 8;
  cvt_bf16_kernel<<<dim3((unsigned)((n8 + 255) / 256)), 256, 0, s>>>(in, out, n8);
}

static inline void gemm(hipStream_t s,
                        const unsigned short* A, long asi, long azs,
                        const unsigned short* B, long bsi, long bzs,
                        const float* bias,
                        const float* resid, long rld, long rzs,
                        float* Cf, long cld, long czs,
                        unsigned short* C16, long c16ld, long c16zs,
                        int M, int N, int K, int Z, int act,
                        const float* pmin = nullptr, const float* pmax = nullptr) {
  dim3 g((N + 63) / 64, (M + 15) / 16, Z);
  dim3 b(32, 1, 1);
  if (N % 16)
    gemm_wmma_bf16<true><<<g, b, 0, s>>>(A, asi, azs, B, bsi, bzs, bias,
                                         resid, rld, rzs, Cf, cld, czs,
                                         C16, c16ld, c16zs, M, N, K, act, pmin, pmax);
  else
    gemm_wmma_bf16<false><<<g, b, 0, s>>>(A, asi, azs, B, bsi, bzs, bias,
                                          resid, rld, rzs, Cf, cld, czs,
                                          C16, c16ld, c16zs, M, N, K, act, pmin, pmax);
}

extern "C" void kernel_launch(void* const* d_in, const int* in_sizes, int n_in,
                              void* d_out, int out_size, void* d_ws, size_t ws_size,
                              hipStream_t stream) {
  (void)in_sizes; (void)n_in; (void)out_size; (void)ws_size;
  const float* features = (const float*)d_in[0];
  const float* proj     = (const float*)d_in[1];
  const float* queries0 = (const float*)d_in[3];
  const float* pmin     = (const float*)d_in[4];
  const float* pmax     = (const float*)d_in[5];
  const float* motion_w1 = (const float*)d_in[6];
  const float* motion_b1 = (const float*)d_in[7];
  const float* motion_w2 = (const float*)d_in[8];
  const float* motion_b2 = (const float*)d_in[9];
  const float* type_w1   = (const float*)d_in[10];
  const float* type_b1   = (const float*)d_in[11];
  const float* type_w2   = (const float*)d_in[12];
  const float* type_b2   = (const float*)d_in[13];
  const float* attr_w1   = (const float*)d_in[14];
  const float* attr_b1   = (const float*)d_in[15];
  const float* attr_w2   = (const float*)d_in[16];
  const float* attr_b2   = (const float*)d_in[17];
  const float* fmlp_w1   = (const float*)d_in[18];
  const float* fmlp_b1   = (const float*)d_in[19];
  const float* fmlp_w2   = (const float*)d_in[20];
  const float* fmlp_b2   = (const float*)d_in[21];
  const float* sa_in_w   = (const float*)d_in[22];
  const float* sa_in_b   = (const float*)d_in[23];
  const float* sa_out_w  = (const float*)d_in[24];
  const float* sa_out_b  = (const float*)d_in[25];
  const float* fa_in_w   = (const float*)d_in[26];
  const float* fa_in_b   = (const float*)d_in[27];
  const float* fa_out_w  = (const float*)d_in[28];
  const float* fa_out_b  = (const float*)d_in[29];
  const float* ffn_w1    = (const float*)d_in[30];
  const float* ffn_b1    = (const float*)d_in[31];
  const float* ffn_w2    = (const float*)d_in[32];
  const float* ffn_b2    = (const float*)d_in[33];
  const float* ln1_g = (const float*)d_in[34];
  const float* ln1_b = (const float*)d_in[35];
  const float* ln2_g = (const float*)d_in[36];
  const float* ln2_b = (const float*)d_in[37];
  const float* ln3_g = (const float*)d_in[38];
  const float* ln3_b = (const float*)d_in[39];
  float* out = (float*)d_out;

  // ---------------- workspace layout ----------------
  char* cur = (char*)d_ws;
  auto allocF = [&](long n) { float* p = (float*)cur; cur += n * sizeof(float); return p; };
  auto allocH = [&](long n) { unsigned short* p = (unsigned short*)cur; cur += n * sizeof(unsigned short); return p; };

  // fp32 buffers
  float* q   = allocF((long)kNQ * kHD);
  float* x   = allocF((long)kNQ * kHD);
  float* scr = allocF((long)kNH * kNQ * kNQ);
  float* pts = allocF((long)kNQ * kP * 3);
  // bf16 activation buffers
  unsigned short* q16    = allocH((long)kNQ * kHD);
  unsigned short* x16    = allocH((long)kNQ * kHD);
  unsigned short* hid16  = allocH((long)kNQ * 4 * kHD);
  unsigned short* qkv16  = allocH((long)kNQ * 3 * kHD);
  unsigned short* vT16   = allocH((long)kHD * kNQ);
  unsigned short* scr16  = allocH((long)kNH * kNQ * kNQ);
  unsigned short* attn16 = allocH((long)kNQ * kHD);
  unsigned short* feath16= allocH((long)kNQ * kHD);
  unsigned short* agg16  = allocH((long)kNQ * kFD);
  // bf16 weight buffers
  unsigned short* mw1 = allocH((long)kHD * kHD);
  unsigned short* mw2 = allocH(11L * kHD);
  unsigned short* tw1 = allocH((long)kHD * kHD);
  unsigned short* tw2 = allocH(10L * kHD);
  unsigned short* aw1 = allocH((long)kHD * kHD);
  unsigned short* aw2 = allocH(3L * kHD);
  unsigned short* fw1 = allocH((long)kHD * kFD);
  unsigned short* fw2 = allocH((long)kHD * kHD);
  unsigned short* saw16 = allocH((long)kNL * 3 * kHD * kHD);
  unsigned short* sao16 = allocH((long)kNL * kHD * kHD);
  unsigned short* faw16 = allocH((long)kNL * 3 * kHD * kHD);
  unsigned short* fao16 = allocH((long)kNL * kHD * kHD);
  unsigned short* ff1_16 = allocH((long)kNL * 4 * kHD * kHD);
  unsigned short* ff2_16 = allocH((long)kNL * 4 * kHD * kHD);

  // ---------------- one-time (per launch) weight conversion ----------------
  cvt(stream, motion_w1, mw1, (long)kHD * kHD);
  cvt(stream, motion_w2, mw2, 11L * kHD);
  cvt(stream, type_w1,   tw1, (long)kHD * kHD);
  cvt(stream, type_w2,   tw2, 10L * kHD);
  cvt(stream, attr_w1,   aw1, (long)kHD * kHD);
  cvt(stream, attr_w2,   aw2, 3L * kHD);
  cvt(stream, fmlp_w1,   fw1, (long)kHD * kFD);
  cvt(stream, fmlp_w2,   fw2, (long)kHD * kHD);
  cvt(stream, sa_in_w,   saw16, (long)kNL * 3 * kHD * kHD);
  cvt(stream, sa_out_w,  sao16, (long)kNL * kHD * kHD);
  cvt(stream, fa_in_w,   faw16, (long)kNL * 3 * kHD * kHD);
  cvt(stream, fa_out_w,  fao16, (long)kNL * kHD * kHD);
  cvt(stream, ffn_w1,    ff1_16, (long)kNL * 4 * kHD * kHD);
  cvt(stream, ffn_w2,    ff2_16, (long)kNL * 4 * kHD * kHD);

  hipMemcpyAsync(q, queries0, (size_t)kNQ * kHD * sizeof(float),
                 hipMemcpyDeviceToDevice, stream);
  cvt(stream, queries0, q16, (long)kNQ * kHD);

  const float kScale = 0.125f;  // 64^-0.5
  const long HD3 = 3 * kHD;

  for (int i = 0; i < kNL; ++i) {
    float* outL = out + (long)i * kNQ * kNOUT;

    // ---- prediction heads ----
    gemm(stream, q16, kHD, 0, mw1, kHD, 0, motion_b1, nullptr, 0, 0,
         nullptr, 0, 0, hid16, kHD, 0, kNQ, kHD, kHD, 1, 1);
    gemm(stream, hid16, kHD, 0, mw2, kHD, 0, motion_b2, nullptr, 0, 0,
         outL + 0, kNOUT, 0, nullptr, 0, 0, kNQ, 11, kHD, 1, 2, pmin, pmax);
    gemm(stream, q16, kHD, 0, tw1, kHD, 0, type_b1, nullptr, 0, 0,
         nullptr, 0, 0, hid16, kHD, 0, kNQ, kHD, kHD, 1, 1);
    gemm(stream, hid16, kHD, 0, tw2, kHD, 0, type_b2, nullptr, 0, 0,
         outL + 11, kNOUT, 0, nullptr, 0, 0, kNQ, 10, kHD, 1, 0);
    gemm(stream, q16, kHD, 0, aw1, kHD, 0, attr_b1, nullptr, 0, 0,
         nullptr, 0, 0, hid16, kHD, 0, kNQ, kHD, kHD, 1, 1);
    gemm(stream, hid16, kHD, 0, aw2, kHD, 0, attr_b2, nullptr, 0, 0,
         outL + 21, kNOUT, 0, nullptr, 0, 0, kNQ, 3, kHD, 1, 2);

    if (i == kNL - 1) break;

    // ---- sample box points + gather-aggregate camera features ----
    sample_pts_kernel<<<(kNQ * kP + 255) / 256, 256, 0, stream>>>(outL, pts);
    gather_agg_kernel<<<kNQ, kFD, 0, stream>>>(features, proj, pts, agg16);

    // ---- feature MLP: agg(128x256) -> feath(128x512) ----
    gemm(stream, agg16, kFD, 0, fw1, kFD, 0, fmlp_b1, nullptr, 0, 0,
         nullptr, 0, 0, hid16, kHD, 0, kNQ, kHD, kFD, 1, 1);
    gemm(stream, hid16, kHD, 0, fw2, kHD, 0, fmlp_b2, nullptr, 0, 0,
         nullptr, 0, 0, feath16, kHD, 0, kNQ, kHD, kHD, 1, 0);

    // ---- LN1 ----
    layernorm512<<<kNQ, 256, 0, stream>>>(q, x, x16, ln1_g + (long)i * kHD, ln1_b + (long)i * kHD);

    // ---- self-attention ----
    gemm(stream, x16, kHD, 0, saw16 + (long)i * HD3 * kHD, kHD, 0,
         sa_in_b + (long)i * HD3, nullptr, 0, 0,
         nullptr, 0, 0, qkv16, HD3, 0, kNQ, (int)HD3, kHD, 1, 0);
    vT_kernel<<<(kHD * kNQ + 255) / 256, 256, 0, stream>>>(qkv16, vT16);
    // scores[h] = Q_h * K_h^T
    gemm(stream, qkv16, HD3, kHd, qkv16 + kHD, HD3, kHd, nullptr, nullptr, 0, 0,
         scr, kNQ, (long)kNQ * kNQ, nullptr, 0, 0, kNQ, kNQ, kHd, kNH, 0);
    softmax128<<<kNH * kNQ, kNQ, 0, stream>>>(scr, scr16, kScale);
    // O[h] = softmax * V_h   (B = vT16, per-head offset 64 rows)
    gemm(stream, scr16, kNQ, (long)kNQ * kNQ, vT16, kNQ, (long)kHd * kNQ,
         nullptr, nullptr, 0, 0, nullptr, 0, 0, attn16, kHD, kHd,
         kNQ, kHd, kNQ, kNH, 0);
    // out projection + residual
    gemm(stream, attn16, kHD, 0, sao16 + (long)i * kHD * kHD, kHD, 0,
         sa_out_b + (long)i * kHD, x, kHD, 0, x, kHD, 0, nullptr, 0, 0,
         kNQ, kHD, kHD, 1, 0);

    // ---- LN2 ----
    layernorm512<<<kNQ, 256, 0, stream>>>(x, x, x16, ln2_g + (long)i * kHD, ln2_b + (long)i * kHD);

    // ---- cross-attention (Q from x, K/V from feath) ----
    const unsigned short* faw = faw16 + (long)i * HD3 * kHD;
    const float* fab = fa_in_b + (long)i * HD3;
    gemm(stream, x16, kHD, 0, faw, kHD, 0, fab, nullptr, 0, 0,
         nullptr, 0, 0, qkv16, HD3, 0, kNQ, kHD, kHD, 1, 0);
    gemm(stream, feath16, kHD, 0, faw + (long)kHD * kHD, kHD, 0, fab + kHD,
         nullptr, 0, 0, nullptr, 0, 0, qkv16 + kHD, HD3, 0, kNQ, kHD, kHD, 1, 0);
    gemm(stream, feath16, kHD, 0, faw + (long)2 * kHD * kHD, kHD, 0, fab + 2 * kHD,
         nullptr, 0, 0, nullptr, 0, 0, qkv16 + 2 * kHD, HD3, 0, kNQ, kHD, kHD, 1, 0);
    vT_kernel<<<(kHD * kNQ + 255) / 256, 256, 0, stream>>>(qkv16, vT16);
    gemm(stream, qkv16, HD3, kHd, qkv16 + kHD, HD3, kHd, nullptr, nullptr, 0, 0,
         scr, kNQ, (long)kNQ * kNQ, nullptr, 0, 0, kNQ, kNQ, kHd, kNH, 0);
    softmax128<<<kNH * kNQ, kNQ, 0, stream>>>(scr, scr16, kScale);
    gemm(stream, scr16, kNQ, (long)kNQ * kNQ, vT16, kNQ, (long)kHd * kNQ,
         nullptr, nullptr, 0, 0, nullptr, 0, 0, attn16, kHD, kHd,
         kNQ, kHd, kNQ, kNH, 0);
    gemm(stream, attn16, kHD, 0, fao16 + (long)i * kHD * kHD, kHD, 0,
         fa_out_b + (long)i * kHD, x, kHD, 0, x, kHD, 0, nullptr, 0, 0,
         kNQ, kHD, kHD, 1, 0);

    // ---- LN3 ----
    layernorm512<<<kNQ, 256, 0, stream>>>(x, x, x16, ln3_g + (long)i * kHD, ln3_b + (long)i * kHD);

    // ---- FFN (512 -> 2048 -> 512) + residual, result -> q / q16 ----
    gemm(stream, x16, kHD, 0, ff1_16 + (long)i * 4 * kHD * kHD, kHD, 0,
         ffn_b1 + (long)i * 4 * kHD, nullptr, 0, 0,
         nullptr, 0, 0, hid16, 4 * kHD, 0, kNQ, 4 * kHD, kHD, 1, 1);
    gemm(stream, hid16, 4 * kHD, 0, ff2_16 + (long)i * 4 * kHD * kHD, 4 * kHD, 0,
         ffn_b2 + (long)i * kHD, x, kHD, 0, q, kHD, 0, q16, kHD, 0,
         kNQ, kHD, 4 * kHD, 1, 0);
  }
}
